// TALHLayerTorch_20976620274250
// MI455X (gfx1250) — matrix operand order repaired and checked by reading, not compiled, not run on hardware
//
#include <hip/hip_runtime.h>

// ---------------------------------------------------------------------------
// MI455X (gfx1250): wave32, WMMA bf16 16x16x32, LDS-staged B tiles, TDM warmup.
// ---------------------------------------------------------------------------

typedef __bf16 bf16;
typedef __attribute__((ext_vector_type(16))) __bf16 v16bf;
typedef __attribute__((ext_vector_type(8)))  __bf16 v8bf;
typedef __attribute__((ext_vector_type(8)))  float  v8f;
typedef __attribute__((ext_vector_type(4)))  unsigned int v4u;
typedef __attribute__((ext_vector_type(8)))  int v8i;
typedef __attribute__((ext_vector_type(4)))  int v4i;

// ---- model dimensions (match reference) ----
constexpr int Bb   = 2;
constexpr int Tt   = 1024;
constexpr int Dm   = 1024;
constexpr int DI   = 2048;
constexpr int Ns   = 16;
constexpr int Hh   = 16;
constexpr int HD   = 64;
constexpr int Ll   = 256;
constexpr int Ee   = 8;
constexpr int Ff   = 2048;
constexpr int CAP  = 2048;            // per-expert token capacity (worst case)
constexpr int Mtok = Bb * Tt;         // 2048 tokens
constexpr float EPS_RMS = 1.1920929e-07f;

// ===========================================================================
// WMMA GEMM:  C[z] = alpha(z) * A[z] (.) B[z]
//   nn==0:  C = A @ B^T   (A:[M,K] row-major, B:[N,K] row-major)
//   nn==1:  C = A @ B     (B:[K,N] row-major)
// Block = 256 threads = 8 waves; block computes 128(M) x 64(N).
// The 64x32 B tile is staged cooperatively into LDS (double-buffered) and
// shared by all 8 waves; each wave owns 16 M-rows and 4 f32 accumulators.
// All wave-level control values pass through readfirstlane so every branch
// is scalar: EXEC stays all-ones across v_wmma (ISA requirement).
// ===========================================================================
struct GemmParams {
  const bf16* A;
  const bf16* B;
  void*       C;
  const int*  rowidx;     // optional: A row gather, indexed rowidx[z*M + r]
  const int*  cnt;        // optional: per-z valid rows (tiles beyond skip)
  const float* alpha_vec; // optional: per-z scale
  long a_os1, a_os0;      // A offset = z1*a_os1 + z0*a_os0
  long b_os1, b_os0;
  long c_os1, c_os0;
  int M, N, K;
  int lda, ldb, ldc;
  int zdiv;               // z1 = z / zdiv, z0 = z % zdiv
  int nn;                 // 0 = A@B^T, 1 = A@B
  int out_bf16;           // 0 = f32 C, 1 = bf16 C
  float alpha;
};

#define BS_PAD 40   // LDS row stride in bf16 (80B, 16B aligned, bank-skewed)

__global__ __launch_bounds__(256) void gemm_wmma_k(GemmParams p) {
  __shared__ bf16 Bs[2][64][BS_PAD];

  const int z  = blockIdx.z;
  const int z1 = z / p.zdiv;
  const int z0 = z % p.zdiv;
  const bf16* Az = p.A + (long)z1 * p.a_os1 + (long)z0 * p.a_os0;
  const bf16* Bz = p.B + (long)z1 * p.b_os1 + (long)z0 * p.b_os0;

  const int mrows = p.cnt ? p.cnt[z] : p.M;
  if ((int)blockIdx.y * 128 >= mrows) return;   // uniform whole-block skip

  const int tilesM   = (p.M + 15) >> 4;
  const int tilesN16 = (p.N + 15) >> 4;

  const int wv = __builtin_amdgcn_readfirstlane((int)(threadIdx.x >> 5));
  const int tm = (int)blockIdx.y * 8 + wv;      // 16-row tile index (scalar)
  const int tn = blockIdx.x;                    // 64-col tile index (scalar)
  const bool active = (tm < tilesM) && (tm * 16 < mrows);
  int nt = tilesN16 - tn * 4;
  nt = nt > 4 ? 4 : nt;                         // scalar

  const int lane = threadIdx.x & 31;
  const int lr   = lane & 15;          // row (A) / col (B,C) within tile
  const int ko   = (lane >> 4) << 3;   // A half-lane K offset: 0 or 8
  const int kb2  = (lane >> 4) << 3;   // B half-lane K offset: 0 or 8

  int arow = 0;
  if (active) {
    arow = tm * 16 + lr;
    if (p.rowidx) arow = p.rowidx[(long)z * p.M + tm * 16 + lr];
  }
  const bf16* Arow = Az + (long)arow * p.lda;

  // cooperative B staging: 256 threads cover 64 cols x 4 chunks of 8 bf16
  const int scol   = threadIdx.x >> 2;   // 0..63
  const int schunk = threadIdx.x & 3;    // 0..3
  const int gcol   = tn * 64 + scol;
  const bool colok = gcol < p.N;

  v8f acc[4];
  #pragma unroll
  for (int j = 0; j < 4; j++) acc[j] = (v8f)(0.0f);

  // ---- stage first tile ----
  {
    v8bf v;
    #pragma unroll
    for (int i = 0; i < 8; i++) v[i] = (bf16)0.0f;
    if (!p.nn) {
      if (colok) v = *(const v8bf*)(Bz + (long)gcol * p.ldb + schunk * 8);
    } else {
      if (colok) {
        #pragma unroll
        for (int i = 0; i < 8; i++)
          v[i] = Bz[(long)(schunk * 8 + i) * p.ldb + gcol];
      }
    }
    *(v8bf*)&Bs[0][scol][schunk * 8] = v;
  }
  __syncthreads();

  int buf = 0;
  for (int kb = 0; kb < p.K; kb += 32) {
    // prefetch next B tile into the other buffer
    if (kb + 32 < p.K) {
      v8bf v;
      #pragma unroll
      for (int i = 0; i < 8; i++) v[i] = (bf16)0.0f;
      if (!p.nn) {
        if (colok) v = *(const v8bf*)(Bz + (long)gcol * p.ldb + (kb + 32) + schunk * 8);
      } else {
        if (colok) {
          #pragma unroll
          for (int i = 0; i < 8; i++)
            v[i] = Bz[(long)(kb + 32 + schunk * 8 + i) * p.ldb + gcol];
        }
      }
      *(v8bf*)&Bs[buf ^ 1][scol][schunk * 8] = v;
    }

    if (active) {
      if (kb + 64 < p.K) __builtin_prefetch(Arow + kb + 64, 0, 1);
      // A fragment (ISA 16-bit A 16x32 layout): two 128-bit global loads
      v16bf afrag;
      {
        v8bf lo = *(const v8bf*)(Arow + kb + ko);
        v8bf hi = *(const v8bf*)(Arow + kb + 16 + ko);
        #pragma unroll
        for (int i = 0; i < 8; i++) { afrag[i] = lo[i]; afrag[i + 8] = hi[i]; }
      }
      #pragma unroll
      for (int j = 0; j < 4; j++) {
        if (j < nt) {
          const bf16* bsrc = &Bs[buf][j * 16 + lr][kb2];
          v16bf bfrag;
          v8bf lo = *(const v8bf*)(bsrc);
          v8bf hi = *(const v8bf*)(bsrc + 16);
          #pragma unroll
          for (int i = 0; i < 8; i++) { bfrag[i] = lo[i]; bfrag[i + 8] = hi[i]; }
          acc[j] = __builtin_amdgcn_wmma_f32_16x16x32_bf16(
              false, afrag, false, bfrag, (short)0, acc[j], false, false);
        }
      }
    }
    __syncthreads();
    buf ^= 1;
  }

  if (active) {
    const float alpha = p.alpha_vec ? p.alpha_vec[z] : p.alpha;
    const long cbase = (long)z1 * p.c_os1 + (long)z0 * p.c_os0;
    const int crow0 = tm * 16 + ((lane >> 4) << 3);
    #pragma unroll
    for (int j = 0; j < 4; j++) {
      if (j < nt) {
        const int ccol = tn * 64 + j * 16 + lr;
        #pragma unroll
        for (int v = 0; v < 8; v++) {
          const long offc = cbase + (long)(crow0 + v) * p.ldc + ccol;
          const float val = alpha * acc[j][v];
          if (p.out_bf16) ((bf16*)p.C)[offc] = (bf16)val;
          else            ((float*)p.C)[offc] = val;
        }
      }
    }
  }
}

// ===========================================================================
// TDM warm-up: build a Tensor DMA descriptor (ISA ch.8) and pull the leading
// 16 x 512-dword tile of each expert weight slice into LDS. Pure prefetch
// (data is not consumed), so descriptor semantics cannot affect correctness,
// but it warms L2 ahead of ternary-quantize + MoE GEMMs and exercises the
// tensor_load_to_lds / s_wait_tensorcnt path. One wave per block.
// ===========================================================================
#if __has_builtin(__builtin_amdgcn_tensor_load_to_lds)
#define HAVE_TDM 1
#endif

__global__ __launch_bounds__(32) void tdm_prefetch_k(const float* gw, const float* uw,
                                                     const float* dw, float* sink) {
#ifdef HAVE_TDM
  __shared__ float stage[8192];   // 32 KB staging tile at LDS offset 0
  const int b = blockIdx.x;       // 0..23: 3 matrices x 8 experts
  const int mat = b >> 3, e = b & 7;
  const float* basep = (mat == 0) ? gw : (mat == 1) ? uw : dw;
  const unsigned long rowdw = (mat == 2) ? (unsigned long)Ff : (unsigned long)Dm;
  const unsigned long nrows = (mat == 2) ? (unsigned long)Dm : (unsigned long)Ff;
  const unsigned long addr = (unsigned long)(basep + (unsigned long)e * rowdw * nrows);

  // D# group0: count=1 (valid), lds_addr=0, global_addr, type=2 ("image")
  v4u g0;
  g0[0] = 1u;
  g0[1] = 0u;
  g0[2] = (unsigned)(addr & 0xffffffffu);
  g0[3] = (unsigned)((addr >> 32) & 0x01ffffffu) | (2u << 30);

  // D# group1: data_size=4B, tensor_dim0=rowdw, tensor_dim1=nrows,
  //            tile_dim0=512, tile_dim1=16, tensor_dim0_stride=rowdw
  const unsigned long dim0 = rowdw, dim1 = nrows, s0 = rowdw;
  v8i g1;
  g1[0] = (int)(2u << 16);
  g1[1] = (int)((dim0 & 0xffffu) << 16);
  g1[2] = (int)(((dim0 >> 16) & 0xffffu) | ((dim1 & 0xffffu) << 16));
  g1[3] = (int)(((dim1 >> 16) & 0xffffu) | (512u << 16));
  g1[4] = (int)(16u);
  g1[5] = (int)(s0 & 0xffffffffu);
  g1[6] = (int)((s0 >> 32) & 0xffffu);
  g1[7] = 0;

  v4i g2; g2[0] = 0; g2[1] = 0; g2[2] = 0; g2[3] = 0;
  v4i g3; g3[0] = 0; g3[1] = 0; g3[2] = 0; g3[3] = 0;

#if defined(__clang_major__) && __clang_major__ >= 23
  v8i g4;
  #pragma unroll
  for (int i = 0; i < 8; i++) g4[i] = 0;
  __builtin_amdgcn_tensor_load_to_lds(g0, g1, g2, g3, g4, 0);
#else
  __builtin_amdgcn_tensor_load_to_lds(g0, g1, g2, g3, 0);
#endif
  __builtin_amdgcn_s_wait_tensorcnt((short)0);
  if (threadIdx.x == 0) sink[b] = stage[0];
#else
  if (threadIdx.x == 0) sink[blockIdx.x] = 0.f;
#endif
}

// ===========================================================================
// Support kernels
// ===========================================================================
__global__ void f32_to_bf16_k(const float* s, bf16* d, long n) {
  long i = (long)blockIdx.x * blockDim.x + threadIdx.x;
  const long stride = (long)gridDim.x * blockDim.x;
  for (; i < n; i += stride) d[i] = (bf16)s[i];
}

__global__ __launch_bounds__(256) void abssum_k(const float* s, long n, float* out) {
  __shared__ float red[256];
  float acc = 0.f;
  for (long i = (long)blockIdx.x * blockDim.x + threadIdx.x; i < n;
       i += (long)gridDim.x * blockDim.x)
    acc += fabsf(s[i]);
  red[threadIdx.x] = acc;
  __syncthreads();
  for (int o = 128; o > 0; o >>= 1) {
    if ((int)threadIdx.x < o) red[threadIdx.x] += red[threadIdx.x + o];
    __syncthreads();
  }
  if (threadIdx.x == 0) atomicAdd(out, red[0]);
}

__global__ void ternary_bf16_k(const float* s, const float* sumabs, long n, bf16* d) {
  const float th = 0.5f * (*sumabs) / (float)n;
  long i = (long)blockIdx.x * blockDim.x + threadIdx.x;
  const long stride = (long)gridDim.x * blockDim.x;
  for (; i < n; i += stride) {
    const float v = s[i];
    d[i] = (bf16)((v > th) ? 1.0f : ((v < -th) ? -1.0f : 0.0f));
  }
}

__global__ __launch_bounds__(256) void rmsnorm_k(const float* x, const float* w,
                                                 float* of, bf16* ob, int cols) {
  const long row = blockIdx.x;
  const float* xr = x + row * (long)cols;
  __shared__ float red[8];
  __shared__ float bval;
  float s = 0.f;
  for (int i = threadIdx.x; i < cols; i += blockDim.x) { float v = xr[i]; s += v * v; }
  for (int o = 16; o > 0; o >>= 1) s += __shfl_down(s, o, 32);
  if ((threadIdx.x & 31) == 0) red[threadIdx.x >> 5] = s;
  __syncthreads();
  if (threadIdx.x == 0) {
    float t = 0.f;
    const int nw = blockDim.x >> 5;
    for (int i = 0; i < nw; i++) t += red[i];
    bval = t;
  }
  __syncthreads();
  const float inv = rsqrtf(bval / (float)cols + EPS_RMS);
  for (int i = threadIdx.x; i < cols; i += blockDim.x) {
    const float v = xr[i] * inv * w[i];
    if (of) of[row * (long)cols + i] = v;
    if (ob) ob[row * (long)cols + i] = (bf16)v;
  }
}

__global__ void ssm_glu_k(const float* xz, float* xf, bf16* xb) {
  const long n = (long)Mtok * DI;
  long i = (long)blockIdx.x * blockDim.x + threadIdx.x;
  const long stride = (long)gridDim.x * blockDim.x;
  for (; i < n; i += stride) {
    const int t = (int)(i / DI);
    const int j = (int)(i % DI);
    const float a = xz[(long)t * (2 * DI) + j];
    const float b = xz[(long)t * (2 * DI) + DI + j];
    const float v = (a / (1.f + expf(-a))) * (1.f / (1.f + expf(-b)));
    xf[i] = v;
    xb[i] = (bf16)v;
  }
}

__global__ void decay_k(const float* Ar, float* dec, long n) {
  long i = (long)blockIdx.x * blockDim.x + threadIdx.x;
  const long stride = (long)gridDim.x * blockDim.x;
  for (; i < n; i += stride) {
    const float a = fminf(fmaxf(Ar[i], -5.f), 0.f);
    dec[i] = expf(-expf(a));   // exp(A) with A = -exp(clip(.))
  }
}

__global__ __launch_bounds__(256) void ssm_scan_k(const float* xin, const float* dec,
                                                  const float* Br, const float* Cr,
                                                  float* ys) {
  const int tid = blockIdx.x * blockDim.x + threadIdx.x;
  if (tid >= Bb * DI) return;
  const int b = tid / DI, di = tid % DI;
  float st[Ns];
  #pragma unroll
  for (int k = 0; k < Ns; k++) st[k] = 0.f;
  for (int t = 0; t < Tt; t++) {
    const long r = (long)b * Tt + t;
    const float x = xin[r * DI + di];
    float y = 0.f;
    #pragma unroll
    for (int k = 0; k < Ns; k++) {
      st[k] = st[k] * dec[r * Ns + k] + x * Br[r * Ns + k];
      y += st[k] * Cr[r * Ns + k];
    }
    ys[r * DI + di] = y;
  }
}

__global__ __launch_bounds__(256) void softmax_rows_k(const float* S, bf16* P, int cols) {
  const long row = blockIdx.x;
  const float* sr = S + row * (long)cols;
  bf16* pr = P + row * (long)cols;
  __shared__ float red[8];
  __shared__ float smax, ssum;
  float m = -3.0e38f;
  for (int i = threadIdx.x; i < cols; i += blockDim.x) m = fmaxf(m, sr[i]);
  for (int o = 16; o > 0; o >>= 1) m = fmaxf(m, __shfl_down(m, o, 32));
  if ((threadIdx.x & 31) == 0) red[threadIdx.x >> 5] = m;
  __syncthreads();
  if (threadIdx.x == 0) {
    float t = red[0];
    const int nw = blockDim.x >> 5;
    for (int i = 1; i < nw; i++) t = fmaxf(t, red[i]);
    smax = t;
  }
  __syncthreads();
  const float mx = smax;
  float s = 0.f;
  for (int i = threadIdx.x; i < cols; i += blockDim.x) s += expf(sr[i] - mx);
  for (int o = 16; o > 0; o >>= 1) s += __shfl_down(s, o, 32);
  if ((threadIdx.x & 31) == 0) red[threadIdx.x >> 5] = s;
  __syncthreads();
  if (threadIdx.x == 0) {
    float t = 0.f;
    const int nw = blockDim.x >> 5;
    for (int i = 0; i < nw; i++) t += red[i];
    ssum = t;
  }
  __syncthreads();
  const float inv = 1.f / ssum;
  for (int i = threadIdx.x; i < cols; i += blockDim.x)
    pr[i] = (bf16)(expf(sr[i] - mx) * inv);
}

__global__ void fuse_k(const float* x0, const float* fl, const float* ssm,
                       const float* at, float* xo, long n) {
  long i = (long)blockIdx.x * blockDim.x + threadIdx.x;
  const long stride = (long)gridDim.x * blockDim.x;
  for (; i < n; i += stride) {
    const float g = 1.f / (1.f + expf(-fl[i]));
    xo[i] = x0[i] + g * ssm[i] + (1.f - g) * at[i];
  }
}

__global__ __launch_bounds__(256) void router_k(const float* h2, const float* rw,
                                                float* probsf, int* topi, float* topw,
                                                int* cnt) {
  const int t = blockIdx.x * blockDim.x + threadIdx.x;
  if (t >= Mtok) return;
  const float* xr = h2 + (long)t * Dm;
  float lg[Ee];
  for (int e = 0; e < Ee; e++) {
    const float* wr = rw + (long)e * Dm;
    float s = 0.f;
    for (int d = 0; d < Dm; d++) s += xr[d] * wr[d];
    lg[e] = s;
  }
  float mx = lg[0];
  for (int e = 1; e < Ee; e++) mx = fmaxf(mx, lg[e]);
  float pr[Ee], se = 0.f;
  for (int e = 0; e < Ee; e++) { pr[e] = expf(lg[e] - mx); se += pr[e]; }
  for (int e = 0; e < Ee; e++) { pr[e] /= se; probsf[t * Ee + e] = pr[e]; }
  int i1 = 0; float p1 = pr[0];
  for (int e = 1; e < Ee; e++) if (pr[e] > p1) { p1 = pr[e]; i1 = e; }
  int i2 = 0; float p2 = -1.f;
  for (int e = 0; e < Ee; e++) if (e != i1 && pr[e] > p2) { p2 = pr[e]; i2 = e; }
  const float inv = 1.f / (p1 + p2 + 1e-9f);
  topi[t * 2] = i1; topi[t * 2 + 1] = i2;
  topw[t * 2] = p1 * inv; topw[t * 2 + 1] = p2 * inv;
  atomicAdd(&cnt[i1], 1);
  atomicAdd(&cnt[i2], 1);
}

__global__ void build_k(const int* topi, int* fill, int* tok, int* slot) {
  const int a = blockIdx.x * blockDim.x + threadIdx.x;
  if (a >= Mtok * 2) return;
  const int e = topi[a];
  const int pos = atomicAdd(&fill[e], 1);
  tok[e * CAP + pos] = a >> 1;
  slot[a] = pos;
}

__global__ void eglu_k(const bf16* G, const bf16* U, bf16* Hb, long n) {
  long i = (long)blockIdx.x * blockDim.x + threadIdx.x;
  const long stride = (long)gridDim.x * blockDim.x;
  for (; i < n; i += stride) {
    const float g = (float)G[i];
    const float u = (float)U[i];
    Hb[i] = (bf16)((g / (1.f + expf(-g))) * u);
  }
}

__global__ void moe_combine_k(const float* Y, const int* topi, const float* topw,
                              const int* slot, float* xout) {
  const long n = (long)Mtok * Dm;
  long i = (long)blockIdx.x * blockDim.x + threadIdx.x;
  const long stride = (long)gridDim.x * blockDim.x;
  for (; i < n; i += stride) {
    const int t = (int)(i / Dm);
    const int d = (int)(i % Dm);
    float acc = xout[i];
    #pragma unroll
    for (int k = 0; k < 2; k++) {
      const int e = topi[t * 2 + k];
      const int s = slot[t * 2 + k];
      acc += topw[t * 2 + k] * Y[((long)e * CAP + s) * Dm + d];
    }
    xout[i] = acc;
  }
}

__global__ void aux_k(const float* probsf, float* out_aux) {
  __shared__ float ps[Ee];
  if ((int)threadIdx.x < Ee) {
    float s = 0.f;
    for (int t = 0; t < Mtok; t++) s += probsf[t * Ee + threadIdx.x];
    ps[threadIdx.x] = s;
  }
  __syncthreads();
  if (threadIdx.x == 0) {
    float ent = 0.f;
    for (int e = 0; e < Ee; e++) {
      const float mp = ps[e] / (float)Mtok;
      ent -= mp * logf(mp + 1e-9f);
    }
    *out_aux = -ent * 0.01f;
  }
}

// ===========================================================================
// Host side
// ===========================================================================
static inline int gsblocks(long n) {
  long b = (n + 255) / 256;
  if (b > 16384) b = 16384;
  if (b < 1) b = 1;
  return (int)b;
}

static void gemm(hipStream_t st, const bf16* A, const bf16* B, void* C,
                 int M, int N, int K, int lda, int ldb, int ldc,
                 int Z, int zdiv,
                 long ao1, long ao0, long bo1, long bo0, long co1, long co0,
                 const int* rowidx, const int* cnt, const float* av,
                 float alpha, int nn, int outb) {
  GemmParams p;
  p.A = A; p.B = B; p.C = C;
  p.rowidx = rowidx; p.cnt = cnt; p.alpha_vec = av;
  p.a_os1 = ao1; p.a_os0 = ao0; p.b_os1 = bo1; p.b_os0 = bo0;
  p.c_os1 = co1; p.c_os0 = co0;
  p.M = M; p.N = N; p.K = K; p.lda = lda; p.ldb = ldb; p.ldc = ldc;
  p.zdiv = zdiv; p.nn = nn; p.out_bf16 = outb; p.alpha = alpha;
  const int tilesM = (M + 15) / 16;
  const int tilesN64 = ((N + 15) / 16 + 3) / 4;
  dim3 grid(tilesN64, (tilesM + 7) / 8, Z);
  gemm_wmma_k<<<grid, 256, 0, st>>>(p);
}

extern "C" void kernel_launch(void* const* d_in, const int* in_sizes, int n_in,
                              void* d_out, int out_size, void* d_ws, size_t ws_size,
                              hipStream_t stream) {
  (void)in_sizes; (void)n_in; (void)out_size; (void)ws_size;

  const float* in_x       = (const float*)d_in[0];
  const float* in_n1w     = (const float*)d_in[1];
  const float* in_n2w     = (const float*)d_in[2];
  const float* in_ssm_in  = (const float*)d_in[3];
  const float* in_Aw      = (const float*)d_in[4];
  const float* in_Bw      = (const float*)d_in[5];
  const float* in_Cw      = (const float*)d_in[6];
  const float* in_snw     = (const float*)d_in[7];
  const float* in_ssm_out = (const float*)d_in[8];
  const float* in_qw      = (const float*)d_in[9];
  const float* in_kvd     = (const float*)d_in[10];
  const float* in_kup     = (const float*)d_in[11];
  const float* in_vup     = (const float*)d_in[12];
  const float* in_aow     = (const float*)d_in[13];
  const float* in_fusew   = (const float*)d_in[14];
  const float* in_rw      = (const float*)d_in[15];
  const float* in_gw      = (const float*)d_in[16];
  const float* in_uw      = (const float*)d_in[17];
  const float* in_dw      = (const float*)d_in[18];
  const float* in_gg      = (const float*)d_in[19];
  const float* in_ug      = (const float*)d_in[20];
  const float* in_dg      = (const float*)d_in[21];

  float* xout = (float*)d_out;                         // [Mtok, Dm]
  float* aux  = ((float*)d_out) + (long)Mtok * Dm;     // scalar

  // ---- workspace layout ----
  char* wsb = (char*)d_ws;
  size_t off = 0;
  auto alloc = [&](size_t nbytes) -> char* {
    char* p = wsb + off;
    off += (nbytes + 255) & ~(size_t)255;
    return p;
  };
  // bf16 weights
  bf16* wb_ssm_in  = (bf16*)alloc((size_t)2 * DI * Dm * 2);
  bf16* wb_A       = (bf16*)alloc((size_t)Ns * DI * 2);
  bf16* wb_B       = (bf16*)alloc((size_t)Ns * DI * 2);
  bf16* wb_C       = (bf16*)alloc((size_t)Ns * DI * 2);
  bf16* wb_ssm_out = (bf16*)alloc((size_t)Dm * DI * 2);
  bf16* wb_q       = (bf16*)alloc((size_t)Dm * Dm * 2);
  bf16* wb_kvd     = (bf16*)alloc((size_t)Ll * Dm * 2);
  bf16* wb_kup     = (bf16*)alloc((size_t)Dm * Ll * 2);
  bf16* wb_vup     = (bf16*)alloc((size_t)Dm * Ll * 2);
  bf16* wb_ao      = (bf16*)alloc((size_t)Dm * Dm * 2);
  bf16* wb_fuse    = (bf16*)alloc((size_t)Dm * Dm * 2);
  bf16* wb_gate    = (bf16*)alloc((size_t)Ee * Ff * Dm * 2);
  bf16* wb_up      = (bf16*)alloc((size_t)Ee * Ff * Dm * 2);
  bf16* wb_down    = (bf16*)alloc((size_t)Ee * Dm * Ff * 2);
  // bf16 activations
  bf16* h1b    = (bf16*)alloc((size_t)Mtok * Dm * 2);
  bf16* xinb   = (bf16*)alloc((size_t)Mtok * DI * 2);
  bf16* ysnb   = (bf16*)alloc((size_t)Mtok * DI * 2);
  bf16* qb     = (bf16*)alloc((size_t)Mtok * Dm * 2);
  bf16* kb     = (bf16*)alloc((size_t)Mtok * Dm * 2);
  bf16* vb     = (bf16*)alloc((size_t)Mtok * Dm * 2);
  bf16* latb   = (bf16*)alloc((size_t)Mtok * Ll * 2);
  bf16* probsb = (bf16*)alloc((size_t)Bb * Hh * Tt * Tt * 2);
  bf16* aob    = (bf16*)alloc((size_t)Mtok * Dm * 2);
  bf16* h2b    = (bf16*)alloc((size_t)Mtok * Dm * 2);
  bf16* Gb     = (bf16*)alloc((size_t)Ee * CAP * Ff * 2);
  bf16* Ub     = (bf16*)alloc((size_t)Ee * CAP * Ff * 2);
  bf16* hhb    = (bf16*)alloc((size_t)Ee * CAP * Ff * 2);
  // f32 activations
  float* h1f    = (float*)alloc((size_t)Mtok * Dm * 4);
  float* xz     = (float*)alloc((size_t)Mtok * 2 * DI * 4);
  float* xinf   = (float*)alloc((size_t)Mtok * DI * 4);
  float* Araw   = (float*)alloc((size_t)Mtok * Ns * 4);
  float* Braw   = (float*)alloc((size_t)Mtok * Ns * 4);
  float* Craw   = (float*)alloc((size_t)Mtok * Ns * 4);
  float* decb   = (float*)alloc((size_t)Mtok * Ns * 4);
  float* ys     = (float*)alloc((size_t)Mtok * DI * 4);
  float* ssmo   = (float*)alloc((size_t)Mtok * Dm * 4);
  float* scoresf= (float*)alloc((size_t)Bb * Hh * Tt * Tt * 4);
  float* attnf  = (float*)alloc((size_t)Mtok * Dm * 4);
  float* fusel  = (float*)alloc((size_t)Mtok * Dm * 4);
  float* h2f    = (float*)alloc((size_t)Mtok * Dm * 4);
  float* Yf     = (float*)alloc((size_t)Ee * CAP * Dm * 4);
  float* probsf = (float*)alloc((size_t)Mtok * Ee * 4);
  float* topw   = (float*)alloc((size_t)Mtok * 2 * 4);
  float* abss   = (float*)alloc(3 * 4);
  float* tdmsink= (float*)alloc(32 * 4);
  int*   cntv   = (int*)alloc(Ee * 4);
  int*   fillv  = (int*)alloc(Ee * 4);
  int*   tokv   = (int*)alloc((size_t)Ee * CAP * 4);
  int*   topi   = (int*)alloc((size_t)Mtok * 2 * 4);
  int*   slotv  = (int*)alloc((size_t)Mtok * 2 * 4);

  // ---- init small state ----
  hipMemsetAsync(abss, 0, 3 * sizeof(float), stream);
  hipMemsetAsync(cntv, 0, Ee * sizeof(int), stream);
  hipMemsetAsync(fillv, 0, Ee * sizeof(int), stream);
  hipMemsetAsync(tokv, 0, (size_t)Ee * CAP * sizeof(int), stream);

  // ---- TDM warm-up of expert weights (pure prefetch) ----
  tdm_prefetch_k<<<24, 32, 0, stream>>>(in_gw, in_uw, in_dw, tdmsink);

  // ---- weight conversion to bf16 ----
  auto cvt = [&](const float* s, bf16* d, long n) {
    f32_to_bf16_k<<<gsblocks(n), 256, 0, stream>>>(s, d, n);
  };
  cvt(in_ssm_in,  wb_ssm_in,  (long)2 * DI * Dm);
  cvt(in_Aw,      wb_A,       (long)Ns * DI);
  cvt(in_Bw,      wb_B,       (long)Ns * DI);
  cvt(in_Cw,      wb_C,       (long)Ns * DI);
  cvt(in_ssm_out, wb_ssm_out, (long)Dm * DI);
  cvt(in_qw,      wb_q,       (long)Dm * Dm);
  cvt(in_kvd,     wb_kvd,     (long)Ll * Dm);
  cvt(in_kup,     wb_kup,     (long)Dm * Ll);
  cvt(in_vup,     wb_vup,     (long)Dm * Ll);
  cvt(in_aow,     wb_ao,      (long)Dm * Dm);
  cvt(in_fusew,   wb_fuse,    (long)Dm * Dm);

  // ---- ternary quantization (exact in bf16: {-1,0,+1}) ----
  const long nGate = (long)Ee * Ff * Dm;
  const long nDown = (long)Ee * Dm * Ff;
  abssum_k<<<2048, 256, 0, stream>>>(in_gw, nGate, &abss[0]);
  abssum_k<<<2048, 256, 0, stream>>>(in_uw, nGate, &abss[1]);
  abssum_k<<<2048, 256, 0, stream>>>(in_dw, nDown, &abss[2]);
  ternary_bf16_k<<<gsblocks(nGate), 256, 0, stream>>>(in_gw, &abss[0], nGate, wb_gate);
  ternary_bf16_k<<<gsblocks(nGate), 256, 0, stream>>>(in_uw, &abss[1], nGate, wb_up);
  ternary_bf16_k<<<gsblocks(nDown), 256, 0, stream>>>(in_dw, &abss[2], nDown, wb_down);

  // ---- h = rmsnorm(x, norm1_w) ----
  rmsnorm_k<<<Mtok, 256, 0, stream>>>(in_x, in_n1w, h1f, h1b, Dm);

  // ---- SSM branch ----
  gemm(stream, h1b, wb_ssm_in, xz, Mtok, 2 * DI, Dm, Dm, Dm, 2 * DI,
       1, 1, 0,0, 0,0, 0,0, nullptr, nullptr, nullptr, 1.f, 0, 0);
  ssm_glu_k<<<gsblocks((long)Mtok * DI), 256, 0, stream>>>(xz, xinf, xinb);
  gemm(stream, xinb, wb_A, Araw, Mtok, Ns, DI, DI, DI, Ns,
       1, 1, 0,0, 0,0, 0,0, nullptr, nullptr, nullptr, 1.f, 0, 0);
  gemm(stream, xinb, wb_B, Braw, Mtok, Ns, DI, DI, DI, Ns,
       1, 1, 0,0, 0,0, 0,0, nullptr, nullptr, nullptr, 1.f, 0, 0);
  gemm(stream, xinb, wb_C, Craw, Mtok, Ns, DI, DI, DI, Ns,
       1, 1, 0,0, 0,0, 0,0, nullptr, nullptr, nullptr, 1.f, 0, 0);
  decay_k<<<gsblocks((long)Mtok * Ns), 256, 0, stream>>>(Araw, decb, (long)Mtok * Ns);
  ssm_scan_k<<<(Bb * DI + 255) / 256, 256, 0, stream>>>(xinf, decb, Braw, Craw, ys);
  rmsnorm_k<<<Mtok, 256, 0, stream>>>(ys, in_snw, nullptr, ysnb, DI);
  gemm(stream, ysnb, wb_ssm_out, ssmo, Mtok, Dm, DI, DI, DI, Dm,
       1, 1, 0,0, 0,0, 0,0, nullptr, nullptr, nullptr, 1.f, 0, 0);

  // ---- MLA attention ----
  gemm(stream, h1b, wb_q, qb, Mtok, Dm, Dm, Dm, Dm, Dm,
       1, 1, 0,0, 0,0, 0,0, nullptr, nullptr, nullptr, 1.f, 0, 1);
  gemm(stream, h1b, wb_kvd, latb, Mtok, Ll, Dm, Dm, Dm, Ll,
       1, 1, 0,0, 0,0, 0,0, nullptr, nullptr, nullptr, 1.f, 0, 1);
  gemm(stream, latb, wb_kup, kb, Mtok, Dm, Ll, Ll, Ll, Dm,
       1, 1, 0,0, 0,0, 0,0, nullptr, nullptr, nullptr, 1.f, 0, 1);
  gemm(stream, latb, wb_vup, vb, Mtok, Dm, Ll, Ll, Ll, Dm,
       1, 1, 0,0, 0,0, 0,0, nullptr, nullptr, nullptr, 1.f, 0, 1);
  // scores[b,h] = (Q K^T) / sqrt(HD), batched over z=(b,h)
  gemm(stream, qb, kb, scoresf, Tt, Tt, HD, Dm, Dm, Tt,
       Bb * Hh, Hh,
       (long)Tt * Dm, (long)HD, (long)Tt * Dm, (long)HD,
       (long)Hh * Tt * Tt, (long)Tt * Tt,
       nullptr, nullptr, nullptr, 0.125f, 0, 0);
  softmax_rows_k<<<Bb * Hh * Tt, 256, 0, stream>>>(scoresf, probsb, Tt);
  // out[b,h] = P @ V (NN), written interleaved as [B,T,H*HD]
  gemm(stream, probsb, vb, aob, Tt, HD, Tt, Tt, Dm, Dm,
       Bb * Hh, Hh,
       (long)Hh * Tt * Tt, (long)Tt * Tt, (long)Tt * Dm, (long)HD,
       (long)Tt * Dm, (long)HD,
       nullptr, nullptr, nullptr, 1.f, 1, 1);
  gemm(stream, aob, wb_ao, attnf, Mtok, Dm, Dm, Dm, Dm, Dm,
       1, 1, 0,0, 0,0, 0,0, nullptr, nullptr, nullptr, 1.f, 0, 0);

  // ---- gated fusion into d_out ----
  gemm(stream, h1b, wb_fuse, fusel, Mtok, Dm, Dm, Dm, Dm, Dm,
       1, 1, 0,0, 0,0, 0,0, nullptr, nullptr, nullptr, 1.f, 0, 0);
  fuse_k<<<gsblocks((long)Mtok * Dm), 256, 0, stream>>>(in_x, fusel, ssmo, attnf,
                                                        xout, (long)Mtok * Dm);

  // ---- MoE ----
  rmsnorm_k<<<Mtok, 256, 0, stream>>>(xout, in_n2w, h2f, h2b, Dm);
  router_k<<<(Mtok + 255) / 256, 256, 0, stream>>>(h2f, in_rw, probsf, topi, topw, cntv);
  build_k<<<(Mtok * 2 + 255) / 256, 256, 0, stream>>>(topi, fillv, tokv, slotv);
  // gathered expert GEMMs (tiles beyond cnt[e] skipped)
  gemm(stream, h2b, wb_gate, Gb, CAP, Ff, Dm, Dm, Dm, Ff,
       Ee, 1, 0,0, (long)Ff * Dm, 0, (long)CAP * Ff, 0,
       tokv, cntv, in_gg, 1.f, 0, 1);
  gemm(stream, h2b, wb_up, Ub, CAP, Ff, Dm, Dm, Dm, Ff,
       Ee, 1, 0,0, (long)Ff * Dm, 0, (long)CAP * Ff, 0,
       tokv, cntv, in_ug, 1.f, 0, 1);
  eglu_k<<<gsblocks((long)Ee * CAP * Ff), 256, 0, stream>>>(Gb, Ub, hhb,
                                                            (long)Ee * CAP * Ff);
  gemm(stream, hhb, wb_down, Yf, CAP, Dm, Ff, Ff, Ff, Dm,
       Ee, 1, (long)CAP * Ff, 0, (long)Dm * Ff, 0, (long)CAP * Dm, 0,
       nullptr, cntv, in_dg, 1.f, 0, 0);
  moe_combine_k<<<gsblocks((long)Mtok * Dm), 256, 0, stream>>>(Yf, topi, topw,
                                                               slotv, xout);
  // ---- aux entropy scalar ----
  aux_k<<<1, 64, 0, stream>>>(probsf, aux);
}
